// EmergentPatternDetector_48550310314466
// MI455X (gfx1250) — compile-verified, matrix operand order checked by reading, not run on hardware
//
#include <hip/hip_runtime.h>

typedef _Float16 f16_t;
typedef __attribute__((ext_vector_type(16))) _Float16 v16h;
typedef __attribute__((ext_vector_type(8)))  _Float16 v8h;
typedef __attribute__((ext_vector_type(8)))  float    v8f;

#define NTOK   32768      // B*S
#define DIN    1024
#define H1DIM  256
#define H2DIM  128
#define ENCD   64
#define GDIM   128        // 4*LSTM_H
#define LSTMH  32
#define SEQ    2048
#define NBATCH 16
#define MEMN   1000

// ---------------------------------------------------------------- WMMA utils
__device__ __forceinline__ v8f wmma16x16x32(v16h a, v16h b, v8f c) {
  return __builtin_amdgcn_wmma_f32_16x16x32_f16(false, a, false, b,
                                                (short)0, c, false, false);
}

// Load a 16x32 f16 fragment (A-layout; B uses same pattern with transposed
// [N][K] storage). p points at the tile origin, ld = row stride in halves.
// Requires 16-byte alignment of p + m*ld (ld multiple of 8, k0 multiple of 32).
__device__ __forceinline__ v16h load_frag(const f16_t* p, int ld, int k0) {
  int lane = threadIdx.x & 31;
  int m  = lane & 15;
  int kb = k0 + ((lane & 16) ? 8 : 0);
  const v8h* q = (const v8h*)(p + (size_t)m * ld + kb);
  v8h lo = q[0];     // K kb..kb+7
  v8h hi = q[2];     // K kb+16..kb+23
  v16h r;
#pragma unroll
  for (int j = 0; j < 8; ++j) { r[j] = lo[j]; r[j + 8] = hi[j]; }
  return r;
}

// Store a 16x16 f32 accumulator tile to LDS as f16 with bias (+relu)
__device__ __forceinline__ void store_act_f16(f16_t* dst, int ld, int n0,
                                              v8f c, const float* bias) {
  int lane  = threadIdx.x & 31;
  int col   = n0 + (lane & 15);
  int rbase = (lane & 16) ? 8 : 0;
  float bv  = bias[col];
#pragma unroll
  for (int r = 0; r < 8; ++r) {
    float v = c[r] + bv;
    v = v > 0.f ? v : 0.f;                 // relu
    dst[(size_t)(rbase + r) * ld + col] = (f16_t)v;
  }
}

// ---------------------------------------------------------------- kernels
// 1) one-shot weight convert+transpose to f16:  Wt[n][k] = (f16)W[k][n]
__global__ void k_convw(const float* __restrict__ W1, const float* __restrict__ W2,
                        const float* __restrict__ W3, const float* __restrict__ Wih,
                        f16_t* __restrict__ W1t, f16_t* __restrict__ W2t,
                        f16_t* __restrict__ W3t, f16_t* __restrict__ Wiht) {
  int i = blockIdx.x * 256 + threadIdx.x;
  if (i < DIN * H1DIM) {                       // W1 [1024][256] -> W1t [256][1024]
    int k = i / H1DIM, n = i % H1DIM;
    W1t[(size_t)n * DIN + k] = (f16_t)W1[i];
  } else if (i < DIN * H1DIM + H1DIM * H2DIM) { // W2 [256][128] -> W2t [128][256]
    int j = i - DIN * H1DIM;
    int k = j / H2DIM, n = j % H2DIM;
    W2t[(size_t)n * H1DIM + k] = (f16_t)W2[j];
  } else if (i < DIN * H1DIM + H1DIM * H2DIM + H2DIM * ENCD) { // W3 [128][64]
    int j = i - DIN * H1DIM - H1DIM * H2DIM;
    int k = j / ENCD, n = j % ENCD;
    W3t[(size_t)n * H2DIM + k] = (f16_t)W3[j];
  } else if (i < DIN * H1DIM + H1DIM * H2DIM + H2DIM * ENCD + ENCD * GDIM) {
    int j = i - DIN * H1DIM - H1DIM * H2DIM - H2DIM * ENCD;  // Wih [64][128]
    int k = j / GDIM, n = j % GDIM;
    Wiht[(size_t)n * ENCD + k] = (f16_t)Wih[j];
  }
}

// 2) fused 3-layer MLP encoder, one 16-row tile per workgroup, WMMA f16->f32
__global__ __launch_bounds__(256) void
k_mlp(const float* __restrict__ X, const float* __restrict__ b1,
      const float* __restrict__ b2, const float* __restrict__ b3,
      const f16_t* __restrict__ W1t, const f16_t* __restrict__ W2t,
      const f16_t* __restrict__ W3t, float* __restrict__ encOut) {
  __shared__ __align__(16) f16_t Ax[16 * DIN];    // 32 KB
  __shared__ __align__(16) f16_t H1[16 * H1DIM];  //  8 KB
  __shared__ __align__(16) f16_t H2[16 * H2DIM];  //  4 KB
  int tile = blockIdx.x;
  int t = threadIdx.x, lane = t & 31, wave = t >> 5;

  // load+convert X tile (16 x 1024 f32 contiguous)
  const float4* src4 = (const float4*)(X + (size_t)tile * 16 * DIN);
  for (int i = t; i < 16 * DIN / 4; i += 256) {
    float4 v = src4[i];
    int b = i * 4;
    Ax[b]     = (f16_t)v.x; Ax[b + 1] = (f16_t)v.y;
    Ax[b + 2] = (f16_t)v.z; Ax[b + 3] = (f16_t)v.w;
  }
  __syncthreads();

  // layer 1: [16,1024] x [1024,256]; wave handles 2 col-tiles of 16
  {
    int n0 = wave * 32, n1 = n0 + 16;
    v8f c0 = {}, c1 = {};
    const f16_t* B0 = W1t + (size_t)n0 * DIN;
    const f16_t* B1 = W1t + (size_t)n1 * DIN;
    for (int k0 = 0; k0 < DIN; k0 += 32) {
      v16h a  = load_frag(Ax, DIN, k0);
      v16h w0 = load_frag(B0, DIN, k0);
      v16h w1 = load_frag(B1, DIN, k0);
      c0 = wmma16x16x32(a, w0, c0);
      c1 = wmma16x16x32(a, w1, c1);
    }
    store_act_f16(H1, H1DIM, n0, c0, b1);
    store_act_f16(H1, H1DIM, n1, c1, b1);
  }
  __syncthreads();

  // layer 2: [16,256] x [256,128]; wave handles 1 col-tile
  {
    int n0 = wave * 16;
    v8f c = {};
    const f16_t* Bw = W2t + (size_t)n0 * H1DIM;
    for (int k0 = 0; k0 < H1DIM; k0 += 32) {
      v16h a = load_frag(H1, H1DIM, k0);
      v16h w = load_frag(Bw, H1DIM, k0);
      c = wmma16x16x32(a, w, c);
    }
    store_act_f16(H2, H2DIM, n0, c, b2);
  }
  __syncthreads();

  // layer 3: [16,128] x [128,64]; waves 0..3, f32 out (no relu)
  if (wave < 4) {
    int n0 = wave * 16;
    v8f c = {};
    const f16_t* Bw = W3t + (size_t)n0 * H2DIM;
    for (int k0 = 0; k0 < H2DIM; k0 += 32) {
      v16h a = load_frag(H2, H2DIM, k0);
      v16h w = load_frag(Bw, H2DIM, k0);
      c = wmma16x16x32(a, w, c);
    }
    int col = n0 + (lane & 15);
    int rbase = (lane & 16) ? 8 : 0;
    float bv = b3[col];
    float* dst = encOut + (size_t)tile * 16 * ENCD;
#pragma unroll
    for (int r = 0; r < 8; ++r)
      dst[(size_t)(rbase + r) * ENCD + col] = c[r] + bv;
  }
}

// 3) G = enc @ Wih + (bih+bhh)   [32768,128], WMMA
__global__ __launch_bounds__(256) void
k_gemm_g(const float* __restrict__ enc, const f16_t* __restrict__ Wiht,
         const float* __restrict__ bih, const float* __restrict__ bhh,
         float* __restrict__ G) {
  __shared__ __align__(16) f16_t Ae[16 * ENCD];
  int tile = blockIdx.x;
  int t = threadIdx.x, lane = t & 31, wave = t >> 5;
  const float* src = enc + (size_t)tile * 16 * ENCD;
  for (int i = t; i < 16 * ENCD; i += 256) Ae[i] = (f16_t)src[i];
  __syncthreads();
  int n0 = wave * 16;              // 8 waves x 16 cols = 128
  v8f c = {};
  const f16_t* Bw = Wiht + (size_t)n0 * ENCD;
  for (int k0 = 0; k0 < ENCD; k0 += 32) {
    v16h a = load_frag(Ae, ENCD, k0);
    v16h w = load_frag(Bw, ENCD, k0);
    c = wmma16x16x32(a, w, c);
  }
  int col = n0 + (lane & 15);
  int rbase = (lane & 16) ? 8 : 0;
  float bv = bih[col] + bhh[col];
#pragma unroll
  for (int r = 0; r < 8; ++r)
    G[((size_t)tile * 16 + rbase + r) * GDIM + col] = c[r] + bv;
}

// 4) LSTM recurrence + emergence classifier; one WG per batch element
__global__ __launch_bounds__(128) void
k_lstm(const float* __restrict__ G, const float* __restrict__ Whh,
       const float* __restrict__ Wc, const float* __restrict__ bc,
       float* __restrict__ emOut) {
  __shared__ float sW[LSTMH * GDIM];   // 16 KB
  __shared__ float sh[LSTMH], sc[LSTMH], sg[GDIM], sWc[LSTMH];
  int b = blockIdx.x, j = threadIdx.x;
  for (int i = j; i < LSTMH * GDIM; i += 128) sW[i] = Whh[i];
  if (j < LSTMH) { sWc[j] = Wc[j]; sh[j] = 0.f; sc[j] = 0.f; }
  float bcv = bc[0];
  __syncthreads();
  for (int tstep = 0; tstep < SEQ; ++tstep) {
    size_t row = (size_t)b * SEQ + tstep;
    float g = G[row * GDIM + j];
#pragma unroll
    for (int k = 0; k < LSTMH; ++k) g += sh[k] * sW[k * GDIM + j];
    float act = (j >= 64 && j < 96) ? tanhf(g) : 1.f / (1.f + expf(-g));
    sg[j] = act;
    __syncthreads();
    if (j < LSTMH) {                    // exactly wave 0
      float cn = sg[32 + j] * sc[j] + sg[j] * sg[64 + j];
      float hn = sg[96 + j] * tanhf(cn);
      sc[j] = cn; sh[j] = hn;
      float p = hn * sWc[j];
#pragma unroll
      for (int off = 16; off; off >>= 1) p += __shfl_xor(p, off, 32);
      if (j == 0) emOut[row] = 1.f / (1.f + expf(-(p + bcv)));
    }
    __syncthreads();
  }
}

// 5) per-row L2 norms of enc
__global__ void k_norms(const float* __restrict__ enc, float* __restrict__ nrm) {
  int n = blockIdx.x * 256 + threadIdx.x;
  if (n < NTOK) {
    const float* p = enc + (size_t)n * ENCD;
    float s = 0.f;
#pragma unroll
    for (int k = 0; k < ENCD; ++k) s += p[k] * p[k];
    nrm[n] = sqrtf(s);
  }
}

// 6) sequential novelty scan; memory ring-buffer lives entirely in 320KB LDS
__global__ __launch_bounds__(1024) void
k_novelty(const float* __restrict__ enc, const float* __restrict__ nrm,
          float* __restrict__ novOut) {
  extern __shared__ float smem[];
  float* mem = smem;                    // [1000][64]
  float* mn  = mem + MEMN * ENCD;       // [1000]
  float* xs  = mn + MEMN;               // [64]
  float* red = xs + ENCD;               // [32]
  __shared__ int s_count, s_ptr, s_ins, s_insptr;
  __shared__ float s_xn;
  int t = threadIdx.x, lane = t & 31, wave = t >> 5;
  if (t == 0) { s_count = 0; s_ptr = 0; }
  __syncthreads();
  for (int n = 0; n < NTOK; ++n) {
    if (t < ENCD) xs[t] = enc[(size_t)n * ENCD + t];
    if (t == 0)   s_xn = nrm[n];
    __syncthreads();
    int cnt = s_count;
    float best = -1e30f;
    if (t < MEMN && t < cnt) {
      const float4* r4 = (const float4*)(mem + (size_t)t * ENCD);
      const float4* x4 = (const float4*)xs;
      float d = 0.f;
#pragma unroll
      for (int k = 0; k < ENCD / 4; ++k) {
        float4 a = r4[k], x = x4[k];
        d += a.x * x.x + a.y * x.y + a.z * x.z + a.w * x.w;
      }
      best = d / (s_xn * mn[t] + 1e-8f);
    }
#pragma unroll
    for (int off = 16; off; off >>= 1) {
      float o = __shfl_xor(best, off, 32);
      best = o > best ? o : best;
    }
    if (lane == 0) red[wave] = best;
    __syncthreads();
    if (wave == 0) {
      float v = red[lane];
#pragma unroll
      for (int off = 16; off; off >>= 1) {
        float o = __shfl_xor(v, off, 32);
        v = o > v ? o : v;
      }
      if (lane == 0) {
        float nv = (cnt == 0) ? 1.0f : (1.0f - v);
        novOut[n] = nv;
        int ins = nv > 0.8f;
        s_ins = ins; s_insptr = s_ptr;
        if (ins) {
          s_ptr = (s_ptr + 1) % MEMN;
          s_count = cnt < MEMN ? cnt + 1 : MEMN;
        }
      }
    }
    __syncthreads();
    if (s_ins) {
      if (t < ENCD) mem[(size_t)s_insptr * ENCD + t] = xs[t];
      if (t == ENCD) mn[s_insptr] = s_xn;
    }
    __syncthreads();
  }
}

// 7) combined threshold
__global__ void k_combine(const float* __restrict__ em, const float* __restrict__ nv,
                          float* __restrict__ ev) {
  int n = blockIdx.x * 256 + threadIdx.x;
  if (n < NTOK) {
    float c = 0.7f * em[n] + 0.3f * nv[n];
    ev[n] = c > 0.8f ? 1.f : 0.f;
  }
}

// ---------------------------------------------------------------- launch
extern "C" void kernel_launch(void* const* d_in, const int* in_sizes, int n_in,
                              void* d_out, int out_size, void* d_ws, size_t ws_size,
                              hipStream_t stream) {
  (void)in_sizes; (void)n_in; (void)out_size; (void)ws_size;
  const float* X   = (const float*)d_in[0];
  const float* W1  = (const float*)d_in[1];
  const float* b1  = (const float*)d_in[2];
  const float* W2  = (const float*)d_in[3];
  const float* b2  = (const float*)d_in[4];
  const float* W3  = (const float*)d_in[5];
  const float* b3  = (const float*)d_in[6];
  const float* Wih = (const float*)d_in[7];
  const float* Whh = (const float*)d_in[8];
  const float* bih = (const float*)d_in[9];
  const float* bhh = (const float*)d_in[10];
  const float* Wc  = (const float*)d_in[11];
  const float* bc  = (const float*)d_in[12];

  float* out = (float*)d_out;
  float* em  = out;                 // [B,S,1]
  float* nov = out + NTOK;          // [B,S]
  float* ev  = out + 2 * NTOK;      // [B,S]
  float* enc = out + 3 * NTOK;      // [B,S,64]

  char* ws = (char*)d_ws;
  f16_t* W1t  = (f16_t*)(ws);                         // 512 KB
  f16_t* W2t  = (f16_t*)(ws + 524288);                //  64 KB
  f16_t* W3t  = (f16_t*)(ws + 589824);                //  16 KB
  f16_t* Wiht = (f16_t*)(ws + 606208);                //  16 KB
  float* G    = (float*)(ws + 622592);                //  16 MB
  float* nrm  = (float*)(ws + 622592 + 16777216);     // 128 KB

  k_convw  <<<dim3(1216), dim3(256), 0, stream>>>(W1, W2, W3, Wih, W1t, W2t, W3t, Wiht);
  k_mlp    <<<dim3(NTOK / 16), dim3(256), 0, stream>>>(X, b1, b2, b3, W1t, W2t, W3t, enc);
  k_norms  <<<dim3(NTOK / 256), dim3(256), 0, stream>>>(enc, nrm);
  k_gemm_g <<<dim3(NTOK / 16), dim3(256), 0, stream>>>(enc, Wiht, bih, bhh, G);
  k_lstm   <<<dim3(NBATCH), dim3(128), 0, stream>>>(G, Whh, Wc, bc, em);
  size_t novShm = (size_t)(MEMN * ENCD + MEMN + ENCD + 32) * sizeof(float);
  k_novelty<<<dim3(1), dim3(1024), novShm, stream>>>(enc, nrm, nov);
  k_combine<<<dim3(NTOK / 256), dim3(256), 0, stream>>>(em, nov, ev);
}
